// SRUCell_23630910063184
// MI455X (gfx1250) — matrix operand order repaired and checked by reading, not compiled, not run on hardware
//
#include <hip/hip_runtime.h>
#include <hip/hip_bf16.h>

// ---------------------------------------------------------------------------
// SRU cell, L=1024 B=32 D=1024.
//  Kernel 1: u = x(32768x1024) @ W(1024x3072), bf16 WMMA, f32 accumulate -> d_ws
//            256x64 block tile, ping-pong LDS, 8 WMMAs / wave / k-step.
//  Kernel 2: sequential scan over t, parallel over B*D lanes.
// ---------------------------------------------------------------------------

typedef __attribute__((ext_vector_type(16))) __bf16 v16bf;
typedef __attribute__((ext_vector_type(8)))  __bf16 v8bf;
typedef __attribute__((ext_vector_type(4)))  __bf16 v4bf;
typedef __attribute__((ext_vector_type(8)))  float  v8f;
typedef __attribute__((ext_vector_type(4)))  float  v4f;

#define L_DIM 1024
#define B_DIM 32
#define D_DIM 1024
#define N_DIM 3072              // 3*D
#define M_DIM 32768             // L*B

#define MT 256                  // block M tile
#define NT 64                   // block N tile
#define KT 32                   // k step = WMMA K
#define NK (D_DIM / KT)         // 32 k-steps
#define ASTR 40                 // LDS row stride (bf16): 80B, 16B-aligned, conflict-free
#define BSTR 40

// ---------------------------------------------------------------------------
// GEMM: 256 threads = 8 waves. Wave w owns M rows [w*32, w*32+32) x all 64 N
// columns: 2 A-fragments x 4 B-fragments = 8 WMMAs per k-step, each B
// fragment reused by two consecutive WMMAs. Ping-pong LDS: global loads for
// tile k+1 issue before the WMMA block, converts/ds_stores after it, so
// s_wait_loadcnt lands behind the matrix ops.
// ---------------------------------------------------------------------------
__global__ __launch_bounds__(256) void sru_gemm(const float* __restrict__ X,
                                                const float* __restrict__ W,
                                                float* __restrict__ U)
{
    __shared__ __align__(16) __bf16 lA[2][MT * ASTR];   // [row][k]  2 x 20480 B
    __shared__ __align__(16) __bf16 lB[2][NT * BSTR];   // [n][k]    2 x  5120 B

    const int tid  = threadIdx.x;
    const int wave = tid >> 5;
    const int lane = tid & 31;
    const int rowBase = blockIdx.x * MT;
    const int colBase = blockIdx.y * NT;

    v8f acc[2][4];
    #pragma unroll
    for (int m = 0; m < 2; ++m)
        #pragma unroll
        for (int s = 0; s < 4; ++s) acc[m][s] = 0;

    // staging maps (all dims divide evenly: no bounds checks, EXEC stays all-1)
    const int aCol  = (tid & 7) * 4;    // 0..28, float4 along K
    const int aRow0 = tid >> 3;         // 0..31, +32 per pass (8 passes for 256 rows)
    const int bN    = (tid & 15) * 4;   // 0..60, float4 along N
    const int bK0   = tid >> 4;         // 0..15, +16 per pass (2 passes)

    const int lrow  = lane & 15;
    const int khalf = lane >> 4;        // 0 or 1

    v4f ax[8];                          // staged A tile (fp32, in regs)
    v4f bx[2];                          // staged B tile

    auto globalLoad = [&](int k0) {
        #pragma unroll
        for (int p = 0; p < 8; ++p) {
            const int r = aRow0 + p * 32;
            ax[p] = *reinterpret_cast<const v4f*>(
                        &X[(size_t)(rowBase + r) * D_DIM + k0 + aCol]);
        }
        #pragma unroll
        for (int p = 0; p < 2; ++p) {
            const int k = bK0 + p * 16;
            bx[p] = *reinterpret_cast<const v4f*>(
                        &W[(size_t)(k0 + k) * N_DIM + colBase + bN]);
        }
    };

    auto ldsStore = [&](int buf) {
        #pragma unroll
        for (int p = 0; p < 8; ++p) {
            const int r = aRow0 + p * 32;
            v4bf bv;
            bv[0] = (__bf16)ax[p][0]; bv[1] = (__bf16)ax[p][1];
            bv[2] = (__bf16)ax[p][2]; bv[3] = (__bf16)ax[p][3];
            *reinterpret_cast<v4bf*>(&lA[buf][r * ASTR + aCol]) = bv;  // ds_store_b64
        }
        #pragma unroll
        for (int p = 0; p < 2; ++p) {
            const int k = bK0 + p * 16;                // transpose on store
            lB[buf][(bN + 0) * BSTR + k] = (__bf16)bx[p][0];
            lB[buf][(bN + 1) * BSTR + k] = (__bf16)bx[p][1];
            lB[buf][(bN + 2) * BSTR + k] = (__bf16)bx[p][2];
            lB[buf][(bN + 3) * BSTR + k] = (__bf16)bx[p][3];
        }
    };

    globalLoad(0);
    ldsStore(0);

    for (int kt = 0; kt < NK; ++kt) {
        __syncthreads();                 // staging of buf[kt&1] visible to all waves
        const int cur = kt & 1;

        if (kt + 1 < NK) globalLoad((kt + 1) * KT);   // HBM/L2 latency hides behind WMMAs

        // ---- A fragments (ISA 16-bit A 16x32 layout):
        // lane l: M = l&15; elems 0..7 -> K = khalf*8 + j; 8..15 -> K = 16+khalf*8+j
        v16bf afrag[2];
        #pragma unroll
        for (int m = 0; m < 2; ++m) {
            const __bf16* ap = &lA[cur][(wave * 32 + m * 16 + lrow) * ASTR + khalf * 8];
            v8bf aLo = *reinterpret_cast<const v8bf*>(ap);        // ds_load_b128
            v8bf aHi = *reinterpret_cast<const v8bf*>(ap + 16);
            #pragma unroll
            for (int i = 0; i < 8; ++i) { afrag[m][i] = aLo[i]; afrag[m][8 + i] = aHi[i]; }
        }

        // ---- 4 B fragments, each reused by 2 WMMAs (B: lane l: N=l&15; elems 0..7 -> K = khalf*8 + j; 8..15 -> K = 16+khalf*8+j)
        #pragma unroll
        for (int s = 0; s < 4; ++s) {
            const __bf16* bp = &lB[cur][(s * 16 + lrow) * BSTR + khalf * 8];
            v8bf bLo = *reinterpret_cast<const v8bf*>(bp);
            v8bf bHi = *reinterpret_cast<const v8bf*>(bp + 16);
            v16bf bfrag;
            #pragma unroll
            for (int i = 0; i < 8; ++i) { bfrag[i] = bLo[i]; bfrag[8 + i] = bHi[i]; }

            acc[0][s] = __builtin_amdgcn_wmma_f32_16x16x32_bf16(
                            false, afrag[0], false, bfrag, (short)0, acc[0][s], false, false);
            acc[1][s] = __builtin_amdgcn_wmma_f32_16x16x32_bf16(
                            false, afrag[1], false, bfrag, (short)0, acc[1][s], false, false);
        }

        if (kt + 1 < NK) ldsStore(cur ^ 1);           // fill the other buffer
    }

    // ---- store C (ISA 16x16 f32 C/D layout): VGPR i -> M = i + khalf*8, N = lane&15
    #pragma unroll
    for (int m = 0; m < 2; ++m) {
        const int mBase = rowBase + wave * 32 + m * 16 + khalf * 8;
        #pragma unroll
        for (int s = 0; s < 4; ++s) {
            const int col = colBase + s * 16 + lrow;
            #pragma unroll
            for (int i = 0; i < 8; ++i) {
                U[(size_t)(mBase + i) * N_DIM + col] = acc[m][s][i];
            }
        }
    }
}

// ---------------------------------------------------------------------------
// Scan: one thread per (b,d). Carried dependence is a single FMA; loads of
// u[t] are affine and prefetchable. Lanes in a wave share b -> the len cutoff
// is wave-uniform, so we split the loop and skip transcendentals in the tail.
// ---------------------------------------------------------------------------
__global__ __launch_bounds__(256) void sru_scan(const float* __restrict__ U,
                                                const float* __restrict__ X,
                                                const float* __restrict__ bias,
                                                const float* __restrict__ c0,
                                                const int*   __restrict__ lens,
                                                float* __restrict__ Hout,
                                                float* __restrict__ Cout)
{
    const int idx = blockIdx.x * 256 + threadIdx.x;   // 0..32767
    const int b = idx >> 10;                          // /D
    const int d = idx & 1023;                         // %D

    float c  = c0[idx];
    const float bf = bias[d];
    const float br = bias[D_DIM + d];
    const int  len = lens[b];                         // wave-uniform

    const float* up = U + (size_t)b * N_DIM + 3 * d;  // stride per t: B*3D
    const float* xp = X + (size_t)b * D_DIM + d;      // stride per t: B*D
    float*       hp = Hout + (size_t)b * D_DIM + d;

    int t = 0;
    for (; t < len; ++t) {
        __builtin_prefetch(up + 8 * (size_t)(B_DIM * N_DIM), 0, 1);
        const float u0 = up[0];
        const float u1 = up[1];
        const float u2 = up[2];
        const float xt = *xp;
        const float g1 = 1.0f / (1.0f + __expf(-(u1 + bf)));
        const float g2 = 1.0f / (1.0f + __expf(-(u2 + br)));
        c = (c - u0) * g1 + u0;                       // carried FMA chain
        const float e2 = __expf(2.0f * c);
        const float th = (e2 - 1.0f) / (e2 + 1.0f);   // tanh(c), off the chain
        *hp = (th - xt) * g2 + xt;
        up += B_DIM * N_DIM;
        xp += B_DIM * D_DIM;
        hp += B_DIM * D_DIM;
    }
    for (; t < L_DIM; ++t) {                          // masked tail: h = 0, c frozen
        *hp = 0.0f;
        hp += B_DIM * D_DIM;
    }
    Cout[idx] = c;
}

// ---------------------------------------------------------------------------
extern "C" void kernel_launch(void* const* d_in, const int* in_sizes, int n_in,
                              void* d_out, int out_size, void* d_ws, size_t ws_size,
                              hipStream_t stream)
{
    const float* x    = (const float*)d_in[0];   // (L, B, D)
    const float* w    = (const float*)d_in[1];   // (D, 3D)
    const float* bias = (const float*)d_in[2];   // (2D)
    const float* c0   = (const float*)d_in[3];   // (B, D)
    const int*   lens = (const int*)  d_in[4];   // (B)

    float* h_out = (float*)d_out;                          // (L, B, D)
    float* c_out = h_out + (size_t)L_DIM * B_DIM * D_DIM;  // (B, D)
    float* u     = (float*)d_ws;                           // (L*B, 3D) = 384 MB

    dim3 grid_g(M_DIM / MT, N_DIM / NT);                   // (128, 48)
    sru_gemm<<<grid_g, 256, 0, stream>>>(x, w, u);

    sru_scan<<<(B_DIM * D_DIM) / 256, 256, 0, stream>>>(u, x, bias, c0, lens,
                                                        h_out, c_out);
}
